// Sinkhorn_VectorQuantizer_32074815767045
// MI455X (gfx1250) — matrix operand order repaired and checked by reading, not compiled, not run on hardware
//
#include <hip/hip_runtime.h>
#include <hip/hip_bf16.h>
#include <stdint.h>

// ---------------------------------------------------------------------------
// Sinkhorn Vector Quantizer forward, MI455X (gfx1250), wave32 + WMMA bf16.
// N = 32768 tokens, C = 256 dims, K = 1024 codes.
// GEMM epilogue now fuses the first Sinkhorn row-sum (sumK1), removing one
// full 128MB pass over the logits.
// ---------------------------------------------------------------------------

#define N_TOK   32768
#define C_DIM   256
#define K_CODE  1024
#define SPATIAL 4096            // 16*16*16
#define EPS_INV 20.0f           // 1/0.05
#define TEMP_INV 10.0f          // 1/0.1

typedef __attribute__((ext_vector_type(8)))  float   v8f;
typedef __attribute__((ext_vector_type(16))) __bf16  v16bf;

union FragU { uint4 q[2]; v16bf v; };

static __device__ __forceinline__ unsigned short f2bf(float f) {
    unsigned u = __float_as_uint(f);
    u += 0x7FFFu + ((u >> 16) & 1u);          // round to nearest even
    return (unsigned short)(u >> 16);
}

static __device__ __forceinline__ float wredSum(float v) {
    #pragma unroll
    for (int m = 16; m; m >>= 1) v += __shfl_xor(v, m, 32);
    return v;
}

// --------------------------- W -> bf16 -------------------------------------
__global__ void k_wconv(const float* __restrict__ W, unsigned short* __restrict__ wb) {
    int i = blockIdx.x * 256 + threadIdx.x;   // 262144 total
    wb[i] = f2bf(W[i]);
}

// ------------------- normalize z rows, write bf16 zn ------------------------
// z layout [b, ch, s] with b = n>>12, s = n&4095; zn row-major [N_TOK, C_DIM].
__global__ void k_norm(const float* __restrict__ z, unsigned short* __restrict__ zn) {
    int n = blockIdx.x * 256 + threadIdx.x;   // 32768 total
    int b = n >> 12, s = n & 4095;
    const float* zb = z + (size_t)b * C_DIM * SPATIAL + s;
    float ss = 0.f;
    #pragma unroll 8
    for (int ch = 0; ch < C_DIM; ++ch) { float v = zb[(size_t)ch * SPATIAL]; ss += v * v; }
    float inv = 1.f / fmaxf(sqrtf(ss), 1e-12f);
    unsigned short* zr = zn + (size_t)n * C_DIM;
    #pragma unroll 4
    for (int cc = 0; cc < C_DIM / 8; ++cc) {
        alignas(16) unsigned short tmp[8];
        #pragma unroll
        for (int u = 0; u < 8; ++u)
            tmp[u] = f2bf(zb[(size_t)(cc * 8 + u) * SPATIAL] * inv);
        *(uint4*)(zr + cc * 8) = *(const uint4*)tmp;
    }
}

// ----------------------- GEMM: logits = zn @ W^T ---------------------------
// Wave computes a 16(M) x 64(N) block: 4 accumulators, K loop of 8 x (K=32).
// A (16x32 bf16) and B (32x16 bf16) fragments each load as two b128 per lane
// following the CDNA5 16-bit WMMA VGPR striping (row-major memory).
// Epilogue: fused sumK1[k] += sum_rows exp((x-1)*20)  (first Sinkhorn row step).
__global__ void k_gemm(const unsigned short* __restrict__ zn,
                       const unsigned short* __restrict__ wb,
                       float* __restrict__ logits,
                       float* __restrict__ sumK1) {
    __shared__ float colred[8][64];
    const int wave = threadIdx.x >> 5;
    const int lane = threadIdx.x & 31;
    const int r    = lane & 15;
    const int half = lane >> 4;
    const int m0 = (blockIdx.y * 8 + wave) * 16;
    const int n0 = blockIdx.x * 64;

    v8f acc[4];
    #pragma unroll
    for (int t = 0; t < 4; ++t)
        #pragma unroll
        for (int j = 0; j < 8; ++j) acc[t][j] = 0.f;

    #pragma unroll
    for (int k0 = 0; k0 < C_DIM; k0 += 32) {
        // A fragment: row m0+r, K in [k0+8h, k0+8h+8) U [k0+8h+16, k0+8h+24)
        FragU a;
        const unsigned short* ap = zn + (size_t)(m0 + r) * C_DIM + k0 + half * 8;
        a.q[0] = *(const uint4*)ap;
        a.q[1] = *(const uint4*)(ap + 16);
        #pragma unroll
        for (int t = 0; t < 4; ++t) {
            // B fragment: col n0+16t+r, K in [k0+8h, k0+8h+8) U [k0+8h+16, k0+8h+24)
            FragU bm;
            const unsigned short* bp = wb + (size_t)(n0 + t * 16 + r) * C_DIM + k0 + half * 8;
            bm.q[0] = *(const uint4*)bp;
            bm.q[1] = *(const uint4*)(bp + 16);
            acc[t] = __builtin_amdgcn_wmma_f32_16x16x32_bf16(
                false, a.v, false, bm.v, (short)0, acc[t], false, false);
        }
    }
    // C layout: VGPR j -> row m0 + j + 8*half, col n0 + 16t + r
    #pragma unroll
    for (int t = 0; t < 4; ++t)
        #pragma unroll
        for (int j = 0; j < 8; ++j)
            logits[(size_t)(m0 + j + half * 8) * K_CODE + n0 + t * 16 + r] = acc[t][j];

    // ---- fused first row-sum: per-column sum of E = exp((x-1)*20) ----
    float part[4];
    #pragma unroll
    for (int t = 0; t < 4; ++t) {
        float s = 0.f;
        #pragma unroll
        for (int j = 0; j < 8; ++j) s += __expf((acc[t][j] - 1.f) * EPS_INV);
        // lanes r and r+16 hold the same column (different row halves)
        s += __shfl_xor(s, 16, 32);
        part[t] = s;
    }
    if (half == 0) {
        #pragma unroll
        for (int t = 0; t < 4; ++t) colred[wave][t * 16 + r] = part[t];
    }
    __syncthreads();
    // all 8 waves of the block cover the same 64 columns -> one flush per col
    if (threadIdx.x < 64) {
        float s = 0.f;
        #pragma unroll
        for (int w = 0; w < 8; ++w) s += colred[w][threadIdx.x];
        atomicAdd(&sumK1[n0 + threadIdx.x], s);
    }
}

// ---------------- alpha[k] = 1 / (K * sumK[k]) -----------------------------
__global__ void k_alpha(const float* __restrict__ sumK, float* __restrict__ alpha) {
    int k = blockIdx.x * 256 + threadIdx.x;
    alpha[k] = 1.f / ((float)K_CODE * sumK[k]);
}

// -------- fused col-normalize + next row-sum accumulation ------------------
// Per row: beta = 1/(N * sum_k E*alpha); accumulate sumKnext[k] += E*beta.
// Wave-private LDS slices avoid atomic contention; one flush per block.
__global__ void k_colrow(const float* __restrict__ logits,
                         const float* __restrict__ alpha,
                         float* __restrict__ sumKnext) {
    __shared__ float slice[8][K_CODE];
    const int tid = threadIdx.x, wave = tid >> 5, lane = tid & 31;
    #pragma unroll
    for (int j = 0; j < 32; ++j) slice[wave][j * 32 + lane] = 0.f;

    float aReg[32];
    #pragma unroll
    for (int j = 0; j < 32; ++j) aReg[j] = alpha[j * 32 + lane];

    const int row0 = blockIdx.x * 64 + wave * 8;
    for (int rr = 0; rr < 8; ++rr) {
        const float* rp = logits + (size_t)(row0 + rr) * K_CODE;
        float e[32], c = 0.f;
        #pragma unroll
        for (int j = 0; j < 32; ++j) {
            float x = rp[j * 32 + lane];
            e[j] = __expf((x - 1.f) * EPS_INV);
            c += e[j] * aReg[j];
        }
        c = wredSum(c);
        float beta = 1.f / ((float)N_TOK * c);
        #pragma unroll
        for (int j = 0; j < 32; ++j) slice[wave][j * 32 + lane] += e[j] * beta;
    }
    __syncthreads();
    #pragma unroll
    for (int j2 = 0; j2 < 4; ++j2) {
        int k = tid + j2 * 256;
        float s = 0.f;
        #pragma unroll
        for (int w = 0; w < 8; ++w) s += slice[w][k];
        atomicAdd(&sumKnext[k], s);
    }
}

// -------- final: beta3, argmax(q), loss, counts ----------------------------
__global__ void k_final(const float* __restrict__ logits,
                        const float* __restrict__ alpha,
                        int* __restrict__ indices,
                        int* __restrict__ counts,
                        float* __restrict__ lossSum) {
    __shared__ float lred[8];
    const int tid = threadIdx.x, wave = tid >> 5, lane = tid & 31;
    float aReg[32];
    #pragma unroll
    for (int j = 0; j < 32; ++j) aReg[j] = alpha[j * 32 + lane];

    const int row0 = blockIdx.x * 64 + wave * 8;
    float lacc = 0.f;
    for (int rr = 0; rr < 8; ++rr) {
        const int row = row0 + rr;
        const float* rp = logits + (size_t)row * K_CODE;
        float x[32], c = 0.f, sp = 0.f, best = -1e30f;
        int bidx = 0;
        #pragma unroll
        for (int j = 0; j < 32; ++j) {
            x[j] = rp[j * 32 + lane];
            float ev = __expf((x[j] - 1.f) * EPS_INV);
            float v  = ev * aReg[j];
            c  += v;
            sp += __expf(x[j] * TEMP_INV);
            if (v > best) { best = v; bidx = j * 32 + lane; }
        }
        c  = wredSum(c);
        sp = wredSum(sp);
        #pragma unroll
        for (int m = 16; m; m >>= 1) {
            float ob = __shfl_xor(best, m, 32);
            int   oi = __shfl_xor(bidx, m, 32);
            if (ob > best || (ob == best && oi < bidx)) { best = ob; bidx = oi; }
        }
        float beta  = 1.f / ((float)N_TOK * c);
        float invsp = 1.f / sp;
        float l = 0.f;
        #pragma unroll
        for (int j = 0; j < 32; ++j) {
            float ev = __expf((x[j] - 1.f) * EPS_INV);
            float q  = (float)N_TOK * aReg[j] * ev * beta;
            float p  = __expf(x[j] * TEMP_INV) * invsp;
            l += q * __logf(p + 1e-6f);
        }
        l = wredSum(l);
        if (lane == 0) {
            lacc += l;
            indices[row] = bidx;
            atomicAdd(&counts[bidx], 1);
        }
    }
    if (lane == 0) lred[wave] = lacc;
    __syncthreads();
    if (tid == 0) {
        float s = 0.f;
        #pragma unroll
        for (int w = 0; w < 8; ++w) s += lred[w];
        atomicAdd(lossSum, s);
    }
}

// ---------------- gather: z_q[b,c,l,h,w] = W[idx[n], c] --------------------
__global__ void k_gather(const float* __restrict__ W,
                         const int* __restrict__ indices,
                         float* __restrict__ out) {
    int i = blockIdx.x * 256 + threadIdx.x;   // 8388608 total
    int s = i & 4095;
    int c = (i >> 12) & 255;
    int b = i >> 20;
    int n = (b << 12) | s;
    out[i] = W[(size_t)indices[n] * C_DIM + c];
}

// ---------------- tail: loss scalar + unique count -------------------------
__global__ void k_tail(const int* __restrict__ counts,
                       const float* __restrict__ lossSum,
                       float* __restrict__ outTail) {
    __shared__ int sred[32];
    int tid = threadIdx.x;                    // 1024 threads
    int v = counts[tid] > 0 ? 1 : 0;
    #pragma unroll
    for (int m = 16; m; m >>= 1) v += __shfl_xor(v, m, 32);
    if ((tid & 31) == 0) sred[tid >> 5] = v;
    __syncthreads();
    if (tid == 0) {
        int u = 0;
        #pragma unroll
        for (int i = 0; i < 32; ++i) u += sred[i];
        outTail[0] = -lossSum[0] / ((float)N_TOK * (float)K_CODE);
        outTail[1] = (float)u;
    }
}

// ---------------------------------------------------------------------------
extern "C" void kernel_launch(void* const* d_in, const int* in_sizes, int n_in,
                              void* d_out, int out_size, void* d_ws, size_t ws_size,
                              hipStream_t stream) {
    const float* z = (const float*)d_in[0];   // [8,256,16,16,16]
    const float* W = (const float*)d_in[1];   // [1024,256] (unit rows)
    float* out = (float*)d_out;               // z_q (8388608) + loss + unique

    char* ws = (char*)d_ws;
    const size_t OFF_LOGITS = 0;                                  // 134217728 B
    const size_t OFF_ZN     = OFF_LOGITS + (size_t)N_TOK * K_CODE * 4;   // 16 MiB
    const size_t OFF_WB     = OFF_ZN + (size_t)N_TOK * C_DIM * 2;        // 512 KiB
    const size_t OFF_SUMK1  = OFF_WB + (size_t)K_CODE * C_DIM * 2;
    const size_t OFF_SUMK2  = OFF_SUMK1 + 4096;
    const size_t OFF_SUMK3  = OFF_SUMK2 + 4096;
    const size_t OFF_ALPHA  = OFF_SUMK3 + 4096;
    const size_t OFF_COUNTS = OFF_ALPHA + 4096;
    const size_t OFF_LOSS   = OFF_COUNTS + 4096;
    const size_t OFF_IDX    = OFF_LOSS + 4096;

    float* logits = (float*)(ws + OFF_LOGITS);
    unsigned short* zn = (unsigned short*)(ws + OFF_ZN);
    unsigned short* wb = (unsigned short*)(ws + OFF_WB);
    float* sumK1 = (float*)(ws + OFF_SUMK1);
    float* sumK2 = (float*)(ws + OFF_SUMK2);
    float* sumK3 = (float*)(ws + OFF_SUMK3);
    float* alpha = (float*)(ws + OFF_ALPHA);
    int*   counts = (int*)(ws + OFF_COUNTS);
    float* lossSum = (float*)(ws + OFF_LOSS);
    int*   indices = (int*)(ws + OFF_IDX);

    // zero the stats region (sumK1..lossSum)
    hipMemsetAsync(ws + OFF_SUMK1, 0, OFF_IDX - OFF_SUMK1, stream);

    k_wconv<<<K_CODE * C_DIM / 256, 256, 0, stream>>>(W, wb);
    k_norm <<<N_TOK / 256, 256, 0, stream>>>(z, zn);
    // GEMM + fused Sinkhorn row-sum #1
    k_gemm <<<dim3(K_CODE / 64, N_TOK / 128), 256, 0, stream>>>(zn, wb, logits, sumK1);

    k_alpha<<<K_CODE / 256, 256, 0, stream>>>(sumK1, alpha);        // row step 1
    k_colrow<<<N_TOK / 64, 256, 0, stream>>>(logits, alpha, sumK2); // col 1 + row-sum 2
    k_alpha<<<K_CODE / 256, 256, 0, stream>>>(sumK2, alpha);        // row step 2
    k_colrow<<<N_TOK / 64, 256, 0, stream>>>(logits, alpha, sumK3); // col 2 + row-sum 3
    k_alpha<<<K_CODE / 256, 256, 0, stream>>>(sumK3, alpha);        // row step 3
    k_final<<<N_TOK / 64, 256, 0, stream>>>(logits, alpha, indices, counts, lossSum);

    k_gather<<<(N_TOK * C_DIM) / 256, 256, 0, stream>>>(W, indices, out);
    k_tail  <<<1, 1024, 0, stream>>>(counts, lossSum, out + (size_t)N_TOK * C_DIM);
}
